// MultiHeadAttention_56453050139178
// MI455X (gfx1250) — compile-verified, matrix operand order checked by reading, not run on hardware
//
#include <hip/hip_runtime.h>

// ---------------------------------------------------------------------------
// MI455X (gfx1250) fused multi-head attention, f16 WMMA with f32 accumulation.
//   pass 0: one-time f32 -> f16 cast of X (Q,K,V) and all weights
//   pass 1: q/k/v = (X @ W^T + b) -> f16 head-split [B,H,S,64] (q pre-scaled 1/8)
//   pass 1b: V -> V^T [B,H,64,S] (hoists the transpose out of the flash loop)
//   pass 2: flash attention; K and V^T tiles double-buffered via
//           global_load_async_to_lds_b128 (ASYNCcnt), online softmax in regs
//   pass 3: out = ctx @ Wo^T + bo -> f32
// ---------------------------------------------------------------------------

#define DMODEL 1024
#define SLEN   2048
#define NHEAD  16
#define DK     64
#define NT     (SLEN / 64)   // 32 key tiles

typedef __attribute__((ext_vector_type(16))) _Float16 v16h;
typedef __attribute__((ext_vector_type(8)))  _Float16 v8h;
typedef __attribute__((ext_vector_type(8)))  float    v8f;

// Pointee type of the async-copy builtin's pointer params, per hipcc
// diagnostic: "__attribute__((__vector_size__(4 * sizeof(int)))) int"
typedef int v4i_vs __attribute__((vector_size(16)));
typedef __attribute__((address_space(1))) v4i_vs* gptr_b128;
typedef __attribute__((address_space(3))) v4i_vs* lptr_b128;

#if defined(__has_builtin)
#if __has_builtin(__builtin_amdgcn_global_load_async_to_lds_b128)
#define USE_ASYNC_LDS 1
#endif
#endif

__device__ __forceinline__ void async_b128(const _Float16* g, _Float16* l) {
#ifdef USE_ASYNC_LDS
  __builtin_amdgcn_global_load_async_to_lds_b128(
      (gptr_b128)const_cast<_Float16*>(g), (lptr_b128)l, 0, 0);
#else
  *(v8h*)l = *(const v8h*)g;
#endif
}

template <int N>
__device__ __forceinline__ void wait_async() {
#ifdef USE_ASYNC_LDS
#if __has_builtin(__builtin_amdgcn_s_wait_asynccnt)
  __builtin_amdgcn_s_wait_asynccnt(N);
#else
  asm volatile("s_wait_asynccnt %0" ::"i"(N) : "memory");
#endif
#endif
}

__device__ __forceinline__ v8f wmma16(v16h a, v16h b, v8f c) {
  // v_wmma_f32_16x16x32_f16  (neg_a, A, neg_b, B, c_mod, C, reuse_a, reuse_b)
  return __builtin_amdgcn_wmma_f32_16x16x32_f16(false, a, false, b, (short)0, c,
                                                false, false);
}

// A/B fragment, 16-bit, per ISA 7.12.2: this lane holds K = {0..7, 16..23}
// relative to p (caller already applied the +8 half-lane base offset).
__device__ __forceinline__ v16h frag_from_f16(const _Float16* p) {
  v8h lo = *(const v8h*)(p);
  v8h hi = *(const v8h*)(p + 16);
  v16h r;
#pragma unroll
  for (int j = 0; j < 8; ++j) { r[j] = lo[j]; r[j + 8] = hi[j]; }
  return r;
}

// ---------------------------------------------------------------------------
// Pass 0: elementwise f32 -> f16 cast (8 elements / thread; n % 2048 == 0).
// ---------------------------------------------------------------------------
__global__ __launch_bounds__(256) void cast_f16_kernel(
    const float* __restrict__ s, _Float16* __restrict__ d, int n) {
  const int i = (blockIdx.x * 256 + threadIdx.x) * 8;
  if (i >= n) return;
  float4 x = *(const float4*)(s + i);
  float4 y = *(const float4*)(s + i + 4);
  v8h o;
  o[0] = (_Float16)x.x; o[1] = (_Float16)x.y;
  o[2] = (_Float16)x.z; o[3] = (_Float16)x.w;
  o[4] = (_Float16)y.x; o[5] = (_Float16)y.y;
  o[6] = (_Float16)y.z; o[7] = (_Float16)y.w;
  *(v8h*)(d + i) = o;
}

// ---------------------------------------------------------------------------
// Passes 1 & 3: C = A @ W^T + b, all-f16 operands, f32 accumulate.
// Block: 8 waves as 4(M) x 2(N); wave tile 32M x 64N -> block 128M x 128N.
// Inner loop: 12 x b128 loads per 8 WMMAs, zero conversions.
// Grid: (4096/128, 1024/128) = (32, 8).
// mode 0: f16 head-split [B,H,S,64] output, scaled.   mode 1: f32 [M,N] + bias.
// ---------------------------------------------------------------------------
__global__ __launch_bounds__(256) void gemm_f16_kernel(
    const _Float16* __restrict__ A, const _Float16* __restrict__ W,
    const float* __restrict__ bias, void* __restrict__ outp,
    int mode, float scale) {
  const int lane = threadIdx.x & 31;
  const int wave = threadIdx.x >> 5;
  const int ln   = lane & 15;
  const int base = (lane < 16) ? 0 : 8;   // half-lane K offset
  const int m_base = blockIdx.x * 128 + (wave & 3) * 32;
  const int n_base = blockIdx.y * 128 + (wave >> 2) * 64;

  const _Float16* ar0 = A + (size_t)(m_base + ln) * DMODEL + base;
  const _Float16* ar1 = ar0 + (size_t)16 * DMODEL;
  v8f acc[2][4] = {};

  for (int k0 = 0; k0 < DMODEL; k0 += 32) {
    v16h a0 = frag_from_f16(ar0 + k0);
    v16h a1 = frag_from_f16(ar1 + k0);
#pragma unroll
    for (int nt = 0; nt < 4; ++nt) {
      // B[k][n] = W[n][k]  (x @ W^T): lane = column n -> contiguous W row.
      const _Float16* brow =
          W + (size_t)(n_base + nt * 16 + ln) * DMODEL + k0 + base;
      v16h b = frag_from_f16(brow);
      acc[0][nt] = wmma16(a0, b, acc[0][nt]);
      acc[1][nt] = wmma16(a1, b, acc[1][nt]);
    }
  }

  if (mode == 0) {
    _Float16* out = (_Float16*)outp;
#pragma unroll
    for (int nt = 0; nt < 4; ++nt) {
      const int n = n_base + nt * 16 + ln;
      const float bv = bias[n];
      const int h = n >> 6, d = n & 63;
#pragma unroll
      for (int mt = 0; mt < 2; ++mt)
#pragma unroll
        for (int r = 0; r < 8; ++r) {
          const int m  = m_base + mt * 16 + ((lane < 16) ? r : r + 8);
          const int bi = m >> 11, s = m & (SLEN - 1);
          const size_t idx = ((size_t)(bi * NHEAD + h) * SLEN + s) * DK + d;
          out[idx] = (_Float16)((acc[mt][nt][r] + bv) * scale);
        }
    }
  } else {
    float* out = (float*)outp;
#pragma unroll
    for (int nt = 0; nt < 4; ++nt) {
      const int n = n_base + nt * 16 + ln;
      const float bv = bias[n];
#pragma unroll
      for (int mt = 0; mt < 2; ++mt)
#pragma unroll
        for (int r = 0; r < 8; ++r) {
          const int m = m_base + mt * 16 + ((lane < 16) ? r : r + 8);
          out[(size_t)m * DMODEL + n] = acc[mt][nt][r] + bv;
        }
    }
  }
}

// ---------------------------------------------------------------------------
// Pass 1b: V [B,H,S,64] -> V^T [B,H,64,S], LDS-tiled 64x64 transpose.
// Grid (B*H, S/64), 256 threads.
// ---------------------------------------------------------------------------
__global__ __launch_bounds__(256) void transpose_v_kernel(
    const _Float16* __restrict__ vh, _Float16* __restrict__ vt) {
  __shared__ _Float16 t[64 * 72];  // stride 72 keeps 16B align, skews banks
  const int bh  = blockIdx.x;
  const int s0  = blockIdx.y * 64;
  const int row = threadIdx.x >> 2;        // 0..63
  const int c   = (threadIdx.x & 3) * 16;  // 0,16,32,48

  const _Float16* src =
      vh + (size_t)bh * SLEN * DK + (size_t)(s0 + row) * DK + c;
  *(v8h*)&t[row * 72 + c]     = *(const v8h*)src;
  *(v8h*)&t[row * 72 + c + 8] = *(const v8h*)(src + 8);
  __syncthreads();

  v8h o0, o1;
#pragma unroll
  for (int j = 0; j < 8; ++j) {
    o0[j] = t[(c + j) * 72 + row];
    o1[j] = t[(c + j + 8) * 72 + row];
  }
  _Float16* dst = vt + (size_t)bh * DK * SLEN + (size_t)row * SLEN + s0 + c;
  *(v8h*)dst       = o0;
  *(v8h*)(dst + 8) = o1;
}

// ---------------------------------------------------------------------------
// Pass 2: flash attention. Block = (b,h, 128 q-rows); 8 waves x 16 rows.
// K and V^T tiles double-buffered via async global->LDS (4 x b128 per thread
// per tile); s_wait_asynccnt<=4 retires the in-flight tile (in-order done).
// 16 WMMAs per 64-key tile per wave.
// ---------------------------------------------------------------------------
__global__ __launch_bounds__(256) void flash_attn_kernel(
    const _Float16* __restrict__ Qh, const _Float16* __restrict__ Kh,
    const _Float16* __restrict__ VT, _Float16* __restrict__ Ctx) {
  __shared__ _Float16 k_lds[2][64 * 64];   // k[n][d], ping-pong
  __shared__ _Float16 vT_lds[2][64 * 64];  // v^T[d][n], ping-pong
  __shared__ _Float16 p_lds[8 * 16 * 64];  // per-wave P slab (C->A relayout)

  const int tid  = threadIdx.x;
  const int lane = tid & 31;
  const int wave = tid >> 5;
  const int ln   = lane & 15;
  const int base = (lane < 16) ? 0 : 8;
  const int bh   = blockIdx.x;  // b*16 + h
  const int b    = bh >> 4, h = bh & 15;
  const int q0   = blockIdx.y * 128;
  const size_t head_off = (size_t)bh * SLEN * DK;

  // Q fragments (A operand), resident for whole kernel. q pre-scaled by 1/8.
  const _Float16* qrow =
      Qh + head_off + (size_t)(q0 + wave * 16 + ln) * DK + base;
  v16h qf[2];
  qf[0] = frag_from_f16(qrow);
  qf[1] = frag_from_f16(qrow + 32);

  v8f ctx[4] = {};
  float mi[8], li[8];
#pragma unroll
  for (int r = 0; r < 8; ++r) { mi[r] = -__builtin_inff(); li[r] = 0.f; }
  _Float16* pw = p_lds + wave * (16 * 64);

  // per-thread staging coordinates: 64 rows x 4 chunks of 16 halfs (32 B)
  const int sn   = tid >> 2;          // K: key row n; V^T: row d
  const int scol = (tid & 3) * 16;
  const _Float16* kp0  = Kh + head_off + (size_t)sn * DK + scol;
  const _Float16* vtp0 = VT + head_off + (size_t)sn * SLEN + scol;

  auto issue_tile = [&](int tkt, int bi) {
    const _Float16* kp = kp0 + (size_t)tkt * 64 * DK;   // next 64 key rows
    const _Float16* vp = vtp0 + (size_t)tkt * 64;       // next 64 key cols
    async_b128(kp,     &k_lds[bi][sn * 64 + scol]);
    async_b128(kp + 8, &k_lds[bi][sn * 64 + scol + 8]);
    async_b128(vp,     &vT_lds[bi][sn * 64 + scol]);
    async_b128(vp + 8, &vT_lds[bi][sn * 64 + scol + 8]);
  };

  issue_tile(0, 0);  // prologue

  for (int kt = 0; kt < NT; ++kt) {
    const int cur = kt & 1;
    __syncthreads();  // all waves done with buffer cur^1 (tile kt-1)
    if (kt + 1 < NT) {
      issue_tile(kt + 1, cur ^ 1);
      if (kt + 2 < NT) {  // keep L2 a tile ahead of the async engine
        __builtin_prefetch(kp0 + (size_t)(kt + 2) * 64 * DK, 0, 0);
        __builtin_prefetch(vtp0 + (size_t)(kt + 2) * 64, 0, 0);
      }
      wait_async<4>();  // in-order retire: tile kt's 4 loads are done
    } else {
      wait_async<0>();
    }
    __syncthreads();  // buffer cur published to all waves

    // S = q @ k^T : 4 N-tiles (keys), K-dim = d_k = 64 -> 2 steps each.
    v8f sc[4];
#pragma unroll
    for (int nt = 0; nt < 4; ++nt) {
      v8f c0 = {};
#pragma unroll
      for (int kk = 0; kk < 2; ++kk) {
        v16h bf =
            frag_from_f16(&k_lds[cur][(nt * 16 + ln) * 64 + kk * 32 + base]);
        c0 = wmma16(qf[kk], bf, c0);
      }
      sc[nt] = c0;
    }

    // Online softmax. C layout: lane holds 8 rows (M=r or 8+r), column N=ln.
    float mnew[8], alpha[8];
#pragma unroll
    for (int r = 0; r < 8; ++r) {
      float pm = fmaxf(fmaxf(sc[0][r], sc[1][r]), fmaxf(sc[2][r], sc[3][r]));
      pm = fmaxf(pm, __shfl_xor(pm, 1));
      pm = fmaxf(pm, __shfl_xor(pm, 2));
      pm = fmaxf(pm, __shfl_xor(pm, 4));
      pm = fmaxf(pm, __shfl_xor(pm, 8));  // stays within each 16-lane half
      mnew[r]  = fmaxf(mi[r], pm);
      alpha[r] = __expf(mi[r] - mnew[r]);
    }
#pragma unroll
    for (int nt = 0; nt < 4; ++nt)
#pragma unroll
      for (int r = 0; r < 8; ++r) sc[nt][r] = __expf(sc[nt][r] - mnew[r]);
#pragma unroll
    for (int r = 0; r < 8; ++r) {
      float ps = sc[0][r] + sc[1][r] + sc[2][r] + sc[3][r];
      ps += __shfl_xor(ps, 1);
      ps += __shfl_xor(ps, 2);
      ps += __shfl_xor(ps, 4);
      ps += __shfl_xor(ps, 8);
      li[r] = li[r] * alpha[r] + ps;
      mi[r] = mnew[r];
    }
#pragma unroll
    for (int t = 0; t < 4; ++t)
#pragma unroll
      for (int r = 0; r < 8; ++r) ctx[t][r] *= alpha[r];

    // P: C layout -> A layout via this wave's private LDS slab.
#pragma unroll
    for (int t = 0; t < 4; ++t)
#pragma unroll
      for (int r = 0; r < 8; ++r) {
        const int M = (lane < 16) ? r : r + 8;
        pw[M * 64 + t * 16 + ln] = (_Float16)sc[t][r];
      }
    v16h pf[2];
    pf[0] = frag_from_f16(&pw[ln * 64 + base]);
    pf[1] = frag_from_f16(&pw[ln * 64 + 32 + base]);

    // ctx += P @ V  (K-dim = 64 keys -> 2 steps, N = d in 4 tiles)
#pragma unroll
    for (int dt = 0; dt < 4; ++dt)
#pragma unroll
      for (int kk = 0; kk < 2; ++kk) {
        v16h vf =
            frag_from_f16(&vT_lds[cur][(dt * 16 + ln) * 64 + kk * 32 + base]);
        ctx[dt] = wmma16(pf[kk], vf, ctx[dt]);
      }
  }

  // finalize: divide by row sum, store combined-heads layout [B,S,H*64] f16
#pragma unroll
  for (int dt = 0; dt < 4; ++dt) {
    const int col = h * DK + dt * 16 + ln;
#pragma unroll
    for (int r = 0; r < 8; ++r) {
      const int m = q0 + wave * 16 + ((lane < 16) ? r : r + 8);
      const size_t idx = ((size_t)b * SLEN + m) * DMODEL + col;
      Ctx[idx] = (_Float16)(ctx[dt][r] * (1.f / li[r]));
    }
  }
}

// ---------------------------------------------------------------------------
extern "C" void kernel_launch(void* const* d_in, const int* in_sizes, int n_in,
                              void* d_out, int out_size, void* d_ws,
                              size_t ws_size, hipStream_t stream) {
  const float* Q  = (const float*)d_in[0];
  const float* K  = (const float*)d_in[1];
  const float* V  = (const float*)d_in[2];
  const float* Wq = (const float*)d_in[3];
  const float* bq = (const float*)d_in[4];
  const float* Wk = (const float*)d_in[5];
  const float* bk = (const float*)d_in[6];
  const float* Wv = (const float*)d_in[7];
  const float* bv = (const float*)d_in[8];
  const float* Wo = (const float*)d_in[9];
  const float* bo = (const float*)d_in[10];
  float* out = (float*)d_out;

  // workspace layout (bytes), all f16 (72 MB total):
  //  [0,24M)   Qc,Kc,Vc   casts of inputs      (3 x 8 MB)
  //  [24M,32M) Wqc..Woc   casts of weights     (4 x 2 MB)
  //  [32M,64M) qh,kh,vh,ch projections + ctx   (4 x 8 MB)
  //  [64M,72M) vt         V^T [B,H,64,S]       (8 MB)
  char* ws = (char*)d_ws;
  const size_t MB = 1u << 20;
  _Float16* Qc  = (_Float16*)(ws);
  _Float16* Kc  = (_Float16*)(ws + 8 * MB);
  _Float16* Vc  = (_Float16*)(ws + 16 * MB);
  _Float16* Wqc = (_Float16*)(ws + 24 * MB);
  _Float16* Wkc = (_Float16*)(ws + 26 * MB);
  _Float16* Wvc = (_Float16*)(ws + 28 * MB);
  _Float16* Woc = (_Float16*)(ws + 30 * MB);
  _Float16* qh  = (_Float16*)(ws + 32 * MB);
  _Float16* kh  = (_Float16*)(ws + 40 * MB);
  _Float16* vh  = (_Float16*)(ws + 48 * MB);
  _Float16* ch  = (_Float16*)(ws + 56 * MB);
  _Float16* vt  = (_Float16*)(ws + 64 * MB);

  const int NX = 2 * SLEN * DMODEL;   // 4 Mi elements
  const int NW = DMODEL * DMODEL;     // 1 Mi elements
  dim3 blk(256);
  cast_f16_kernel<<<NX / 2048, blk, 0, stream>>>(Q, Qc, NX);
  cast_f16_kernel<<<NX / 2048, blk, 0, stream>>>(K, Kc, NX);
  cast_f16_kernel<<<NX / 2048, blk, 0, stream>>>(V, Vc, NX);
  cast_f16_kernel<<<NW / 2048, blk, 0, stream>>>(Wq, Wqc, NW);
  cast_f16_kernel<<<NW / 2048, blk, 0, stream>>>(Wk, Wkc, NW);
  cast_f16_kernel<<<NW / 2048, blk, 0, stream>>>(Wv, Wvc, NW);
  cast_f16_kernel<<<NW / 2048, blk, 0, stream>>>(Wo, Woc, NW);

  dim3 gp(32, 8);
  // fold 1/sqrt(dk)=1/8 into q
  gemm_f16_kernel<<<gp, blk, 0, stream>>>(Qc, Wqc, bq, qh, 0, 0.125f);
  gemm_f16_kernel<<<gp, blk, 0, stream>>>(Kc, Wkc, bk, kh, 0, 1.0f);
  gemm_f16_kernel<<<gp, blk, 0, stream>>>(Vc, Wvc, bv, vh, 0, 1.0f);

  transpose_v_kernel<<<dim3(32, 32), blk, 0, stream>>>(vh, vt);

  flash_attn_kernel<<<dim3(32, 16), blk, 0, stream>>>(qh, kh, vt, ch);

  gemm_f16_kernel<<<gp, blk, 0, stream>>>(ch, Woc, bo, out, 1, 0.f);
}
